// SimpleLSTM_23235773071457
// MI455X (gfx1250) — compile-verified, matrix operand order checked by reading, not run on hardware
//
#include <hip/hip_runtime.h>

typedef __attribute__((ext_vector_type(16))) _Float16 v16h;
typedef __attribute__((ext_vector_type(8)))  _Float16 v8h;
typedef __attribute__((ext_vector_type(8)))  float    v8f;
typedef __attribute__((ext_vector_type(4)))  float    v4f;

#define T_STEPS 512
#define I_DIM   16
#define H_DIM   32   // 4H = 128 -> 8 tiles of N=16

// Native TRANS-pipe tanh on gfx1250 (v_tanh_f32); guarded fallback via exp.
#if __has_builtin(__builtin_amdgcn_tanhf)
__device__ __forceinline__ float fast_tanh(float x) {
    return __builtin_amdgcn_tanhf(x);
}
#elif __has_builtin(__builtin_amdgcn_tanh_f32)
__device__ __forceinline__ float fast_tanh(float x) {
    return __builtin_amdgcn_tanh_f32(x);
}
#else
__device__ __forceinline__ float fast_tanh(float x) {
    float e = __expf(2.0f * x);
    return (e - 1.0f) * __builtin_amdgcn_rcpf(e + 1.0f);
}
#endif

__device__ __forceinline__ float fast_sigmoid(float x) {
    // sigma(x) = 0.5*tanh(x/2) + 0.5  -> 1 TRANS + 2 VALU
    return fmaf(0.5f, fast_tanh(0.5f * x), 0.5f);
}

__global__ __launch_bounds__(32) void lstm_wmma_kernel(
    const float* __restrict__ x,     // [B, T, I]
    const float* __restrict__ W_ih,  // [4H, I]
    const float* __restrict__ W_hh,  // [4H, H]
    const float* __restrict__ b_ih,  // [4H]
    const float* __restrict__ b_hh,  // [4H]
    const float* __restrict__ W_fc,  // [1, H]
    const float* __restrict__ b_fc,  // [1]
    float* __restrict__ out,         // [B]
    int T)
{
    __shared__ _Float16 hshm[16 * H_DIM]; // 16 batch rows x 32 h values (f16)

    const int lane = threadIdx.x;      // 0..31
    const int r    = lane & 15;        // row / column-in-tile index
    const int hi   = lane >> 4;        // lane half: selects K-group / M-offset
    const int b0   = blockIdx.x * 16;  // batch tile base

    // ---- Preload weight fragments (B-operand layout, f16) ----
    // B fragment (16x16x32_f16): lane L holds output column j = 16n + (L%16).
    //   lanes 0-15  hold K {0..7, 16..23}; lanes 16-31 hold K {8..15, 24..31}.
    // Bias trick: K=16 of the A operand carries a constant 1.0 (lanes<16,
    // element 8), and wih[n] carries (b_ih+b_hh)[j] at the same K=16 slot, so
    // gates = [x|1] @ [W_ih^T; bias] with C starting from inline 0.
    v16h wih[8], whh[8];
#pragma unroll
    for (int n = 0; n < 8; ++n) {
        const int j = 16 * n + r;           // gate column (0..127)
        const float* pi = W_ih + j * I_DIM + hi * 8;
        v4f a0 = *(const v4f*)(pi);
        v4f a1 = *(const v4f*)(pi + 4);
        v16h fi;
#pragma unroll
        for (int k = 0; k < 4; ++k) {
            fi[k]      = (_Float16)a0[k];
            fi[4 + k]  = (_Float16)a1[k];
            fi[8 + k]  = (_Float16)0.0f;
            fi[12 + k] = (_Float16)0.0f;
        }
        fi[8] = (hi == 0) ? (_Float16)(b_ih[j] + b_hh[j]) : (_Float16)0.0f;
        wih[n] = fi;

        const float* ph = W_hh + j * H_DIM + hi * 8;
        v4f h0 = *(const v4f*)(ph);
        v4f h1 = *(const v4f*)(ph + 4);
        v4f h2 = *(const v4f*)(ph + 16);
        v4f h3 = *(const v4f*)(ph + 20);
        v16h fh;
#pragma unroll
        for (int k = 0; k < 4; ++k) {
            fh[k]      = (_Float16)h0[k];
            fh[4 + k]  = (_Float16)h1[k];
            fh[8 + k]  = (_Float16)h2[k];
            fh[12 + k] = (_Float16)h3[k];
        }
        whh[n] = fh;
    }

    // ---- Recurrent state ----
    v16h hfrag;                  // h in A-fragment layout (f16), starts at 0
#pragma unroll
    for (int k = 0; k < 16; ++k) hfrag[k] = (_Float16)0.0f;
    v8f cst[2] = { {}, {} };     // c in C/D layout: [0]=cols 0..15, [1]=cols 16..31

    const float* xbase = x + (size_t)(b0 + r) * (size_t)T * I_DIM + hi * 8;
    const _Float16 kone = (hi == 0) ? (_Float16)1.0f : (_Float16)0.0f;

    for (int t = 0; t < T; ++t) {
        // x_t A-fragment: row = batch row b0+r; K 0..15 = features,
        // K=16 = 1.0 (bias row), K 17..31 = 0
        v4f xa = *(const v4f*)(xbase + (size_t)t * I_DIM);
        v4f xb = *(const v4f*)(xbase + (size_t)t * I_DIM + 4);
        __builtin_prefetch(xbase + (size_t)(t + 1) * I_DIM, 0, 3);
        v16h xf;
#pragma unroll
        for (int k = 0; k < 4; ++k) {
            xf[k]      = (_Float16)xa[k];
            xf[4 + k]  = (_Float16)xb[k];
            xf[8 + k]  = (_Float16)0.0f;
            xf[12 + k] = (_Float16)0.0f;
        }
        xf[8] = kone;

        // Two tile groups: g=0 -> tiles {0,2,4,6} (h cols 0..15),
        //                  g=1 -> tiles {1,3,5,7} (h cols 16..31).
        // Within a group: i=tile 2*0+g, f=2*1+g, g-gate=2*2+g, o=2*3+g.
#pragma unroll
        for (int g = 0; g < 2; ++g) {
            const v8f zeroc = {};
            v8f acc[4];
#pragma unroll
            for (int q = 0; q < 4; ++q)
                acc[q] = __builtin_amdgcn_wmma_f32_16x16x32_f16(
                             false, xf, false, wih[2 * q + g],
                             (short)0, zeroc, false, false);
#pragma unroll
            for (int q = 0; q < 4; ++q)
                acc[q] = __builtin_amdgcn_wmma_f32_16x16x32_f16(
                             false, hfrag, false, whh[2 * q + g],
                             (short)0, acc[q], false, false);

            // Elementwise LSTM cell. C/D layout: VGPR v, lane L ->
            // row M = v + 8*(L/16), col = 16g + (L%16).
#pragma unroll
            for (int v = 0; v < 8; ++v) {
                float cc = fast_sigmoid(acc[1][v]) * cst[g][v]
                         + fast_sigmoid(acc[0][v]) * fast_tanh(acc[2][v]);
                cst[g][v] = cc;
                float hv = fast_sigmoid(acc[3][v]) * fast_tanh(cc);
                hshm[(v + 8 * hi) * H_DIM + 16 * g + r] = (_Float16)hv;
            }
        }
        __syncthreads();

        // Reload h as A-fragment: lane holds row r, K-groups {hi*8..+7} and
        // {16+hi*8..+7} (two 16B LDS loads)
        {
            v8h lo  = *(const v8h*)&hshm[r * H_DIM + hi * 8];
            v8h hiv = *(const v8h*)&hshm[r * H_DIM + 16 + hi * 8];
#pragma unroll
            for (int k = 0; k < 8; ++k) {
                hfrag[k]     = lo[k];
                hfrag[8 + k] = hiv[k];
            }
        }
        __syncthreads();
    }

    // ---- Final FC: out[b] = h_last[b,:] . W_fc + b_fc ----
    if (lane < 16) {
        float s = b_fc[0];
#pragma unroll
        for (int k = 0; k < H_DIM; ++k)
            s += (float)hshm[r * H_DIM + k] * W_fc[k];
        out[b0 + r] = s;
    }
}

extern "C" void kernel_launch(void* const* d_in, const int* in_sizes, int n_in,
                              void* d_out, int out_size, void* d_ws, size_t ws_size,
                              hipStream_t stream) {
    const float* x    = (const float*)d_in[0];
    const float* W_ih = (const float*)d_in[1];
    const float* W_hh = (const float*)d_in[2];
    const float* b_ih = (const float*)d_in[3];
    const float* b_hh = (const float*)d_in[4];
    const float* W_fc = (const float*)d_in[5];
    const float* b_fc = (const float*)d_in[6];
    float* out = (float*)d_out;

    const int T = T_STEPS;
    const int B = in_sizes[0] / (T * I_DIM);   // 8192 for the reference shapes

    dim3 grid(B / 16), block(32);
    lstm_wmma_kernel<<<grid, block, 0, stream>>>(x, W_ih, W_hh, b_ih, b_hh,
                                                 W_fc, b_fc, out, T);
}
